// HeteroGraphSAGE_18734647345201
// MI455X (gfx1250) — matrix operand order repaired and checked by reading, not compiled, not run on hardware
//
#include <hip/hip_runtime.h>

// ---------------------------------------------------------------------------
// HeteroGraphSAGE for MI455X (gfx1250, wave32).
//  * All dense GEMMs via v_wmma_f32_16x16x32_bf16 (fp32 accumulate).
//  * A-tiles: global_load_b128 (explicit addrspace(1)) + packed f32->bf16.
//  * B-tiles: bf16 W^T staged in LDS (288B pitch), double-buffered
//    ds_load_b128 so loads for tile t+1 overlap the WMMA of tile t.
//  * Edge aggregation via per-relation CSR (built once, reused both layers).
// ---------------------------------------------------------------------------

#define AS1 __attribute__((address_space(1)))

typedef __attribute__((ext_vector_type(16))) __bf16 v16bf;
typedef __attribute__((ext_vector_type(8)))  __bf16 v8bf;
typedef __attribute__((ext_vector_type(8)))  float  v8f;
typedef __attribute__((ext_vector_type(8)))  float  v8fv;
typedef __attribute__((ext_vector_type(4)))  float  f4raw;   // trivial, AS-agnostic

#define HDIM 128
#define KPAD 144   // LDS pitch (elements) for staged W^T: 288B stride, 16B aligned

static __device__ __forceinline__ unsigned short f2bf(float f) {
    unsigned int u = __float_as_uint(f);
    u += 0x7FFFu + ((u >> 16) & 1u);        // round-to-nearest-even to bf16
    return (unsigned short)(u >> 16);
}

union BFrag { v16bf v; uint4 q[2]; };

static __device__ __forceinline__ void loadB(BFrag& b, const unsigned short* Wt,
                                             int n, int koff) {
    const uint4* p = (const uint4*)&Wt[n * KPAD + koff];
    b.q[0] = p[0];
    b.q[1] = p[1];
}

// C[M,N] = act( sum_{t<nterms} A_t[M,K] @ W_t[Kw,N]  + bias0 + bias1 )
// K must be a multiple of 32 (A is padded); W rows in [Kw,K) are treated as 0.
// 256 threads = 8 waves; each wave computes a 16-row stripe x NTILES*16 cols.
template <int NTILES>
__global__ __launch_bounds__(256)
void wmma_gemm_kernel(const float* __restrict__ A0, const float* __restrict__ W0,
                      const float* __restrict__ A1, const float* __restrict__ W1,
                      const float* __restrict__ A2, const float* __restrict__ W2,
                      const float* __restrict__ bias0, const float* __restrict__ bias1,
                      float* __restrict__ C,
                      int M, int N, int K, int Kw, int nterms, int do_relu)
{
    __shared__ unsigned short Wt[HDIM * KPAD];   // W^T in bf16: Wt[n*KPAD + k]

    const int tid  = threadIdx.x;
    const int lane = tid & 31;
    const int wave = tid >> 5;
    const int l15  = lane & 15;
    const int half = lane >> 4;

    const int rowBase = (blockIdx.x * 8 + wave) * 16;
    int arow = rowBase + l15;
    if (arow >= M) arow = M - 1;                 // clamp loads; stores are masked

    v8f acc[NTILES];
    #pragma unroll
    for (int t = 0; t < NTILES; ++t) { v8f z = {}; acc[t] = z; }

    const float* Aptr[3] = {A0, A1, A2};
    const float* Wptr[3] = {W0, W1, W2};

    for (int term = 0; term < nterms; ++term) {
        const float* Wg = Wptr[term];
        const float* Ag = Aptr[term];

        // ---- stage W^T (bf16) into LDS, zero-padded to K x NTILES*16 ----
        const int tot = NTILES * 16 * K;
        for (int i = tid; i < tot; i += 256) {
            int n = i / K, k = i - n * K;
            float w = (n < N && k < Kw) ? ((const AS1 float*)Wg)[(size_t)k * N + n] : 0.f;
            Wt[n * KPAD + k] = f2bf(w);
        }
        __syncthreads();

        const float* Arow = Ag + (size_t)arow * K;
        for (int k0 = 0; k0 < K; k0 += 32) {
            // A fragment: lane half=0 -> K k0..k0+15; half=1 -> K k0+16..k0+31
            v8fv f0 = *(const AS1 v8fv*)(Arow + k0 + half * 16);
            v8fv f1 = *(const AS1 v8fv*)(Arow + k0 + half * 16 + 8);
            v8bf lo = __builtin_convertvector(f0, v8bf);
            v8bf hi = __builtin_convertvector(f1, v8bf);
            v16bf a = __builtin_shufflevector(lo, hi, 0, 1, 2, 3, 4, 5, 6, 7,
                                              8, 9, 10, 11, 12, 13, 14, 15);

            // B fragments: lanes 0-15 K=k0..k0+15, lanes 16-31 K=k0+16..k0+31
            const int koff = k0 + half * 16;
            BFrag b[2];
            loadB(b[0], Wt, l15, koff);
            #pragma unroll
            for (int t = 0; t < NTILES; ++t) {
                if (t + 1 < NTILES) loadB(b[(t + 1) & 1], Wt, (t + 1) * 16 + l15, koff);
                acc[t] = __builtin_amdgcn_wmma_f32_16x16x32_bf16(
                    false, a, false, b[t & 1].v, (short)0, acc[t], false, false);
            }
        }
        __syncthreads();
    }

    // ---- bias + activation + store (fast path for full tiles) ----
    const bool fullRows = (rowBase + 16 <= M);
    #pragma unroll
    for (int t = 0; t < NTILES; ++t) {
        const int col = t * 16 + l15;
        float b = 0.f;
        if (col < N) {
            if (bias0) b += ((const AS1 float*)bias0)[col];
            if (bias1) b += ((const AS1 float*)bias1)[col];
        }
        if (fullRows && (t * 16 + 16 <= N)) {
            #pragma unroll
            for (int r = 0; r < 8; ++r) {
                const int row = rowBase + r + half * 8;
                float v = acc[t][r] + b;
                if (do_relu) v = fmaxf(v, 0.f);
                ((AS1 float*)C)[(size_t)row * N + col] = v;
            }
        } else {
            #pragma unroll
            for (int r = 0; r < 8; ++r) {
                const int row = rowBase + r + half * 8;
                float v = acc[t][r] + b;
                if (do_relu) v = fmaxf(v, 0.f);
                if (row < M && col < N) ((AS1 float*)C)[(size_t)row * N + col] = v;
            }
        }
    }
}

// ---------------------- CSR build ----------------------
__global__ void hist_kernel(const int* __restrict__ dst, int E, int* __restrict__ deg) {
    int i = blockIdx.x * blockDim.x + threadIdx.x;
    if (i < E) atomicAdd(&deg[((const AS1 int*)dst)[i]], 1);
}

// single-workgroup exclusive scan (n up to 400k: ~391 chunk iterations, trivial)
__global__ __launch_bounds__(1024)
void scan_kernel(int* __restrict__ deg_and_cursor, int* __restrict__ rowptr, int n)
{
    __shared__ int sh[1024];
    const int tid = threadIdx.x;
    int running = 0;
    for (int base = 0; base < n; base += 1024) {
        const int idx = base + tid;
        const int val = (idx < n) ? ((const AS1 int*)deg_and_cursor)[idx] : 0;
        sh[tid] = val;
        __syncthreads();
        for (int off = 1; off < 1024; off <<= 1) {
            int t = (tid >= off) ? sh[tid - off] : 0;
            __syncthreads();
            sh[tid] += t;
            __syncthreads();
        }
        const int incl  = sh[tid];
        const int total = sh[1023];
        if (idx < n) {
            const int ex = running + incl - val;
            ((AS1 int*)rowptr)[idx] = ex;
            ((AS1 int*)deg_and_cursor)[idx] = ex;    // becomes the fill cursor
        }
        running += total;
        __syncthreads();
    }
    if (tid == 0) ((AS1 int*)rowptr)[n] = running;
}

__global__ void fill_kernel(const int* __restrict__ src, const int* __restrict__ dst,
                            int E, int* __restrict__ cursor, int* __restrict__ csr_src) {
    int i = blockIdx.x * blockDim.x + threadIdx.x;
    if (i < E) {
        int pos = atomicAdd(&cursor[((const AS1 int*)dst)[i]], 1);
        ((AS1 int*)csr_src)[pos] = ((const AS1 int*)src)[i];
    }
}

// ---------------------- mean aggregation: one wave per dst row ----------------------
__global__ __launch_bounds__(256)
void agg_mean_kernel(const float* __restrict__ hsrc, const int* __restrict__ rowptr,
                     const int* __restrict__ csr_src, float* __restrict__ out, int Ndst)
{
    const int gid  = blockIdx.x * 8 + (threadIdx.x >> 5);
    const int lane = threadIdx.x & 31;
    if (gid >= Ndst) return;
    const int beg = ((const AS1 int*)rowptr)[gid];
    const int end = ((const AS1 int*)rowptr)[gid + 1];
    f4raw acc = {0.f, 0.f, 0.f, 0.f};
    for (int e = beg; e < end; ++e) {
        const int s = ((const AS1 int*)csr_src)[e];
        const f4raw v = ((const AS1 f4raw*)(hsrc + (size_t)s * HDIM))[lane];
        acc += v;
    }
    const float inv = 1.f / fmaxf((float)(end - beg), 1.f);
    acc *= inv;
    ((AS1 f4raw*)(out + (size_t)gid * HDIM))[lane] = acc;
}

__global__ void add_mats_kernel(const float* __restrict__ a, const float* __restrict__ b,
                                float* __restrict__ o, int n) {
    int i = blockIdx.x * blockDim.x + threadIdx.x;
    if (i < n) ((AS1 float*)o)[i] = ((const AS1 float*)a)[i] + ((const AS1 float*)b)[i];
}

// pad rows: out[M,Kout] = [ in[M,Kin] | 0 ]
__global__ void pad_rows_kernel(const float* __restrict__ in, float* __restrict__ out,
                                int M, int Kin, int Kout) {
    int i = blockIdx.x * blockDim.x + threadIdx.x;
    if (i < M * Kout) {
        int r = i / Kout, c = i - r * Kout;
        ((AS1 float*)out)[i] = (c < Kin) ? ((const AS1 float*)in)[(size_t)r * Kin + c] : 0.f;
    }
}

// ---------------------------------------------------------------------------
extern "C" void kernel_launch(void* const* d_in, const int* in_sizes, int n_in,
                              void* d_out, int out_size, void* d_ws, size_t ws_size,
                              hipStream_t stream)
{
    (void)in_sizes; (void)n_in; (void)out_size; (void)ws_size;

    const int NT = 400000, NU = 100000, NM = 20000, E = 500000, H = HDIM;

    const float* x_t = (const float*)d_in[0];
    const float* x_u = (const float*)d_in[1];
    const float* x_m = (const float*)d_in[2];
    const int* su2t = (const int*)d_in[3];
    const int* du2t = (const int*)d_in[4];
    const int* st2u = (const int*)d_in[5];
    const int* dt2u = (const int*)d_in[6];
    const int* st2m = (const int*)d_in[7];
    const int* dt2m = (const int*)d_in[8];
    const int* sm2t = (const int*)d_in[9];
    const int* dm2t = (const int*)d_in[10];
    const float* Pt_w = (const float*)d_in[11]; const float* Pt_b = (const float*)d_in[12];
    const float* Pu_w = (const float*)d_in[13]; const float* Pu_b = (const float*)d_in[14];
    const float* Pm_w = (const float*)d_in[15]; const float* Pm_b = (const float*)d_in[16];
    const float* Wl = (const float*)d_in[17];
    const float* bl = (const float*)d_in[18];
    const float* Wr = (const float*)d_in[19];
    const float* Wf1 = (const float*)d_in[20]; const float* bf1 = (const float*)d_in[21];
    const float* Wf2 = (const float*)d_in[22]; const float* bf2 = (const float*)d_in[23];
    const float* Wg1 = (const float*)d_in[24]; const float* bg1 = (const float*)d_in[25];
    const float* Wg2 = (const float*)d_in[26]; const float* bg2 = (const float*)d_in[27];

    auto WLp = [&](int L, int r) { return Wl + (size_t)(L * 4 + r) * H * H; };
    auto BLp = [&](int L, int r) { return bl + (size_t)(L * 4 + r) * H; };
    auto WRp = [&](int L, int r) { return Wr + (size_t)(L * 4 + r) * H * H; };

    // ---- d_out layout (floats): fraud[NT,2], ring[NT,32], h_t, h_u, h_m ----
    float* out = (float*)d_out;
    float* out_fraud = out;
    float* out_ring  = out + (size_t)NT * 2;
    float* out_ht    = out_ring + (size_t)NT * 32;
    float* out_hu    = out_ht + (size_t)NT * H;
    float* out_hm    = out_hu + (size_t)NU * H;

    // ---- workspace bump allocator ----
    char* wsp = (char*)d_ws;
    auto alloc = [&](size_t bytes) -> void* {
        void* p = (void*)wsp;
        wsp += (bytes + 255) & ~(size_t)255;
        return p;
    };
    float* h_t0   = (float*)alloc((size_t)NT * H * 4);
    float* h_t1   = (float*)alloc((size_t)NT * H * 4);
    float* h_u0   = (float*)alloc((size_t)NU * H * 4);
    float* h_u1   = (float*)alloc((size_t)NU * H * 4);
    float* h_m0   = (float*)alloc((size_t)NM * H * 4);
    float* h_m1   = (float*)alloc((size_t)NM * H * 4);
    float* agg_ut = (float*)alloc((size_t)NT * H * 4);
    float* agg_mt = (float*)alloc((size_t)NT * H * 4);
    float* agg_tu = (float*)alloc((size_t)NU * H * 4);
    float* agg_tm = (float*)alloc((size_t)NM * H * 4);
    float* Wsum   = (float*)alloc((size_t)H * H * 4);
    float* x_mp   = (float*)alloc((size_t)NM * 32 * 4);   // x_m zero-padded K=16->32
    int* rp_u2t = (int*)alloc((size_t)(NT + 1) * 4);
    int* cu_u2t = (int*)alloc((size_t)(NT + 1) * 4);
    int* cs_u2t = (int*)alloc((size_t)E * 4);
    int* rp_m2t = (int*)alloc((size_t)(NT + 1) * 4);
    int* cu_m2t = (int*)alloc((size_t)(NT + 1) * 4);
    int* cs_m2t = (int*)alloc((size_t)E * 4);
    int* rp_t2u = (int*)alloc((size_t)(NU + 1) * 4);
    int* cu_t2u = (int*)alloc((size_t)(NU + 1) * 4);
    int* cs_t2u = (int*)alloc((size_t)E * 4);
    int* rp_t2m = (int*)alloc((size_t)(NM + 1) * 4);
    int* cu_t2m = (int*)alloc((size_t)(NM + 1) * 4);
    int* cs_t2m = (int*)alloc((size_t)E * 4);

    const int EB = (E + 255) / 256;

    auto build_csr = [&](const int* src, const int* dst, int Ndst,
                         int* rowptr, int* cursor, int* csr_src) {
        (void)hipMemsetAsync(cursor, 0, (size_t)Ndst * 4, stream);
        hist_kernel<<<EB, 256, 0, stream>>>(dst, E, cursor);
        scan_kernel<<<1, 1024, 0, stream>>>(cursor, rowptr, Ndst);
        fill_kernel<<<EB, 256, 0, stream>>>(src, dst, E, cursor, csr_src);
    };
    build_csr(su2t, du2t, NT, rp_u2t, cu_u2t, cs_u2t);
    build_csr(sm2t, dm2t, NT, rp_m2t, cu_m2t, cs_m2t);
    build_csr(st2u, dt2u, NU, rp_t2u, cu_t2u, cs_t2u);
    build_csr(st2m, dt2m, NM, rp_t2m, cu_t2m, cs_t2m);

    auto gemm = [&](const float* A0, const float* W0, const float* A1, const float* W1,
                    const float* A2, const float* W2, const float* b0, const float* b1,
                    float* C, int M, int N, int K, int Kw, int nterms, int relu) {
        const int blocks = (M + 127) / 128;
        switch ((N + 15) / 16) {
        case 8: wmma_gemm_kernel<8><<<blocks, 256, 0, stream>>>(A0,W0,A1,W1,A2,W2,b0,b1,C,M,N,K,Kw,nterms,relu); break;
        case 4: wmma_gemm_kernel<4><<<blocks, 256, 0, stream>>>(A0,W0,A1,W1,A2,W2,b0,b1,C,M,N,K,Kw,nterms,relu); break;
        case 2: wmma_gemm_kernel<2><<<blocks, 256, 0, stream>>>(A0,W0,A1,W1,A2,W2,b0,b1,C,M,N,K,Kw,nterms,relu); break;
        default: wmma_gemm_kernel<1><<<blocks, 256, 0, stream>>>(A0,W0,A1,W1,A2,W2,b0,b1,C,M,N,K,Kw,nterms,relu); break;
        }
    };

    // ---- input projections (x_m padded to K=32 so the hot loop is branchless) ----
    pad_rows_kernel<<<(NM * 32 + 255) / 256, 256, 0, stream>>>(x_m, x_mp, NM, 16, 32);
    gemm(x_t, Pt_w, nullptr, nullptr, nullptr, nullptr, Pt_b, nullptr, h_t0, NT, H, 64, 64, 1, 0);
    gemm(x_u, Pu_w, nullptr, nullptr, nullptr, nullptr, Pu_b, nullptr, h_u0, NU, H, 32, 32, 1, 0);
    gemm(x_mp, Pm_w, nullptr, nullptr, nullptr, nullptr, Pm_b, nullptr, h_m0, NM, H, 32, 16, 1, 0);

    // ---- two hetero-SAGE layers ----
    float* ht_cur = h_t0; float* hu_cur = h_u0; float* hm_cur = h_m0;
    for (int L = 0; L < 2; ++L) {
        float* ht_nxt = (L == 0) ? h_t1 : out_ht;
        float* hu_nxt = (L == 0) ? h_u1 : out_hu;
        float* hm_nxt = (L == 0) ? h_m1 : out_hm;

        // fuse the two root transforms on h_t: h_t@(Wr0 + Wr3)
        add_mats_kernel<<<(H * H + 255) / 256, 256, 0, stream>>>(WRp(L, 0), WRp(L, 3), Wsum, H * H);

        agg_mean_kernel<<<(NT + 7) / 8, 256, 0, stream>>>(hu_cur, rp_u2t, cs_u2t, agg_ut, NT);
        agg_mean_kernel<<<(NT + 7) / 8, 256, 0, stream>>>(hm_cur, rp_m2t, cs_m2t, agg_mt, NT);
        agg_mean_kernel<<<(NU + 7) / 8, 256, 0, stream>>>(ht_cur, rp_t2u, cs_t2u, agg_tu, NU);
        agg_mean_kernel<<<(NM + 7) / 8, 256, 0, stream>>>(ht_cur, rp_t2m, cs_t2m, agg_tm, NM);

        // o_t = relu( agg_ut@Wl0 + agg_mt@Wl3 + h_t@(Wr0+Wr3) + bl0 + bl3 )
        gemm(agg_ut, WLp(L, 0), agg_mt, WLp(L, 3), ht_cur, Wsum,
             BLp(L, 0), BLp(L, 3), ht_nxt, NT, H, H, H, 3, 1);
        // o_u = relu( agg_tu@Wl1 + h_u@Wr1 + bl1 )
        gemm(agg_tu, WLp(L, 1), hu_cur, WRp(L, 1), nullptr, nullptr,
             BLp(L, 1), nullptr, hu_nxt, NU, H, H, H, 2, 1);
        // o_m = relu( agg_tm@Wl2 + h_m@Wr2 + bl2 )
        gemm(agg_tm, WLp(L, 2), hm_cur, WRp(L, 2), nullptr, nullptr,
             BLp(L, 2), nullptr, hm_nxt, NM, H, H, H, 2, 1);

        ht_cur = ht_nxt; hu_cur = hu_nxt; hm_cur = hm_nxt;
    }

    // ---- heads (z scratch reuses agg_ut: NT*64 <= NT*128) ----
    float* z = agg_ut;
    gemm(ht_cur, Wf1, nullptr, nullptr, nullptr, nullptr, bf1, nullptr, z, NT, 64, 128, 128, 1, 1);
    gemm(z, Wf2, nullptr, nullptr, nullptr, nullptr, bf2, nullptr, out_fraud, NT, 2, 64, 64, 1, 0);
    gemm(ht_cur, Wg1, nullptr, nullptr, nullptr, nullptr, bg1, nullptr, z, NT, 64, 128, 128, 1, 1);
    gemm(z, Wg2, nullptr, nullptr, nullptr, nullptr, bg2, nullptr, out_ring, NT, 32, 64, 64, 1, 0);
}
